// AutoCF_74079595921836
// MI455X (gfx1250) — matrix operand order repaired and checked by reading, not compiled, not run on hardware
//
#include <hip/hip_runtime.h>
#include <hip/hip_bf16.h>

// ---------------------------------------------------------------------------
// AutoCF forward on MI455X (gfx1250, wave32).
//   out = emb0 + gcn(emb0) + gcn(gcn(emb0)) + graph_transformer(...)
// D = 64, HEADS = 4 (dh = 16).
// Q/K/V projections are done PER NODE (not per edge) as a dense GEMM using
// v_wmma_f32_16x16x32_bf16 (bf16 A/B, f32 accumulate). All gathers/scatters
// are L2-resident (node table = 76.8 MB < 192 MB L2).
// The QKV WMMA kernel is persistent: weights are staged to LDS once per block
// and row tiles are grid-strided, amortizing weight staging ~7x.
// ---------------------------------------------------------------------------

#define D 64
#define HEADS 4
#define XS 66  // padded LDS row stride (ushorts) -> odd bank step, no conflicts

typedef __attribute__((ext_vector_type(16))) __bf16 v16bf;
typedef __attribute__((ext_vector_type(8)))  float  v8f;

__device__ __forceinline__ unsigned short f32_to_bf16_rne(float f) {
    unsigned int u = __float_as_uint(f);
    unsigned int r = u + 0x7FFFu + ((u >> 16) & 1u);
    return (unsigned short)(r >> 16);
}

__device__ __forceinline__ unsigned int pack2_bf16(float lo, float hi) {
    return (unsigned int)f32_to_bf16_rne(lo) |
           ((unsigned int)f32_to_bf16_rne(hi) << 16);
}

// out[i] = concat(u, i) as float4 stream
__global__ __launch_bounds__(256) void concat_kernel(
    const float* __restrict__ uE, const float* __restrict__ iE,
    float* __restrict__ out, long nu4, long total4) {
    long i = (long)blockIdx.x * blockDim.x + threadIdx.x;
    if (i >= total4) return;
    const float4* src = (i < nu4) ? (const float4*)uE : (const float4*)(iE) - nu4;
    ((float4*)out)[i] = src[i];
}

// out += a   (float4 stream)
__global__ __launch_bounds__(256) void add_kernel(
    float* __restrict__ out, const float* __restrict__ a, long total4) {
    long i = (long)blockIdx.x * blockDim.x + threadIdx.x;
    if (i >= total4) return;
    float4 o = ((float4*)out)[i];
    float4 x = ((const float4*)a)[i];
    o.x += x.x; o.y += x.y; o.z += x.z; o.w += x.w;
    ((float4*)out)[i] = o;
}

// SpMM: one wave per edge; lane handles 2 dims (32 lanes * 2 = 64 = D).
__global__ __launch_bounds__(256) void spmm_kernel(
    const float* __restrict__ vals, const int* __restrict__ rows,
    const int* __restrict__ cols, const float* __restrict__ in,
    float* __restrict__ out, long E) {
    long e = ((long)blockIdx.x * blockDim.x + threadIdx.x) >> 5;
    int lane = threadIdx.x & 31;
    if (e >= E) return;
    long r = rows[e], c = cols[e];
    float v = vals[e];
    float2 x = *(const float2*)(in + c * D + lane * 2);
    atomicAdd(out + r * D + lane * 2,     v * x.x);
    atomicAdd(out + r * D + lane * 2 + 1, v * x.y);
}

// Dense QKV projection: [N,64] x [64,64] x3 via v_wmma_f32_16x16x32_bf16.
// Persistent blocks: stage all 3 weights in LDS once, then grid-stride over
// 32-row activation tiles. Block = 256 thr = 8 wave32; wave (rt, ct) owns a
// 16x16 output tile per weight; K = 64 -> 2 k-steps of 32.
__global__ __launch_bounds__(256) void qkv_wmma_kernel(
    const float* __restrict__ X,
    const float* __restrict__ qT, const float* __restrict__ kT,
    const float* __restrict__ vT,
    float* __restrict__ Q, float* __restrict__ K, float* __restrict__ V,
    long N, long numTiles) {
    __shared__ unsigned short sX[32 * XS];        // activations, bf16
    __shared__ unsigned short sW[3 * 64 * XS];    // weights, transposed, bf16
    const float* Ws[3] = {qT, kT, vT};
    float*       Os[3] = {Q, K, V};

    const int t = threadIdx.x;

    // Stage weights transposed ONCE: sW[w][n*XS + k] = W[k*64 + n]
    // (global reads coalesced over n; 16-bit LDS scatter over k)
    for (int w = 0; w < 3; ++w) {
        const float* W = Ws[w];
        for (int i = t; i < 64 * 64; i += 256) {
            int k = i >> 6, n = i & 63;
            sW[w * 64 * XS + n * XS + k] = f32_to_bf16_rne(W[k * 64 + n]);
        }
    }
    __syncthreads();

    const int wave = t >> 5, lane = t & 31;
    const int rt = wave >> 2, ct = wave & 3;   // row-tile (0..1), col-tile (0..3)
    const int m16 = lane & 15, hi = lane >> 4; // hi: lane in upper half-wave

    for (long tile = blockIdx.x; tile < numTiles; tile += gridDim.x) {
        const long rowBase = tile * 32;

        // Stage 32 activation rows: float2 load -> packed 2xbf16 -> b32 DS store
        for (int i = t; i < 32 * 32; i += 256) {   // 1024 float2 units
            int r = i >> 5, c2 = i & 31;
            long g = rowBase + r;
            float2 xv = make_float2(0.0f, 0.0f);
            if (g < N) xv = *(const float2*)(X + g * D + c2 * 2);
            *(unsigned int*)&sX[r * XS + c2 * 2] = pack2_bf16(xv.x, xv.y);
        }
        __syncthreads();

        v8f acc0 = {}, acc1 = {}, acc2 = {};
#pragma unroll
        for (int kt = 0; kt < 2; ++kt) {
            // A fragment (16x32 bf16): lane M = m16; VGPR j: K = hi*8 + 2j (+16)
            union { v16bf v; unsigned int d[8]; } a;
            const unsigned short* xrow = &sX[(rt * 16 + m16) * XS + kt * 32];
#pragma unroll
            for (int j = 0; j < 4; ++j) {
                a.d[j]     = *(const unsigned int*)&xrow[hi * 8 + 2 * j];
                a.d[4 + j] = *(const unsigned int*)&xrow[16 + hi * 8 + 2 * j];
            }
            // B fragments (32x16 bf16): lane N = m16; VGPR j: K = hi*8 + 2j (+16)
#pragma unroll
            for (int w = 0; w < 3; ++w) {
                union { v16bf v; unsigned int d[8]; } b;
                const unsigned short* wrow =
                    &sW[w * 64 * XS + (ct * 16 + m16) * XS + kt * 32];
#pragma unroll
                for (int j = 0; j < 4; ++j) {
                    b.d[j]     = *(const unsigned int*)&wrow[hi * 8 + 2 * j];
                    b.d[4 + j] = *(const unsigned int*)&wrow[16 + hi * 8 + 2 * j];
                }
                if (w == 0)
                    acc0 = __builtin_amdgcn_wmma_f32_16x16x32_bf16(
                        false, a.v, false, b.v, (short)0, acc0, false, false);
                else if (w == 1)
                    acc1 = __builtin_amdgcn_wmma_f32_16x16x32_bf16(
                        false, a.v, false, b.v, (short)0, acc1, false, false);
                else
                    acc2 = __builtin_amdgcn_wmma_f32_16x16x32_bf16(
                        false, a.v, false, b.v, (short)0, acc2, false, false);
            }
        }
        // C layout: VGPR r -> M = r + hi*8, N = m16
#pragma unroll
        for (int w = 0; w < 3; ++w) {
            float* O = Os[w];
#pragma unroll
            for (int r = 0; r < 8; ++r) {
                long gr = rowBase + rt * 16 + hi * 8 + r;
                if (gr < N) {
                    float v = (w == 0) ? acc0[r] : (w == 1) ? acc1[r] : acc2[r];
                    O[gr * D + ct * 16 + m16] = v;
                }
            }
        }
        __syncthreads();   // protect sX before next tile's staging
    }
}

// Edge pass A: expAtt[e][h] = exp(clip(q[row]·k[col] per head)), attNorm += .
// Wave per edge; lane d = lane*2; head = lane/8; 8-lane xor-tree reduction.
__global__ __launch_bounds__(256) void att_score_kernel(
    const int* __restrict__ rows, const int* __restrict__ cols,
    const float* __restrict__ Q, const float* __restrict__ K,
    float* __restrict__ attNorm, float* __restrict__ expAtt, long E) {
    long e = ((long)blockIdx.x * blockDim.x + threadIdx.x) >> 5;
    int lane = threadIdx.x & 31;
    if (e >= E) return;
    long r = rows[e], c = cols[e];
    float2 q = *(const float2*)(Q + r * D + lane * 2);
    float2 k = *(const float2*)(K + c * D + lane * 2);
    float p = q.x * k.x + q.y * k.y;
    p += __shfl_xor(p, 1, 8);
    p += __shfl_xor(p, 2, 8);
    p += __shfl_xor(p, 4, 8);
    if ((lane & 7) == 0) {
        int h = lane >> 3;
        float att = fminf(10.0f, fmaxf(-10.0f, p));
        float ea = __expf(att);
        expAtt[e * HEADS + h] = ea;
        atomicAdd(attNorm + r * HEADS + h, ea);
    }
}

// Edge pass B: out[row] += (expAtt / (attNorm[row]+1e-8)) * V[col]
__global__ __launch_bounds__(256) void att_aggr_kernel(
    const int* __restrict__ rows, const int* __restrict__ cols,
    const float* __restrict__ expAtt, const float* __restrict__ attNorm,
    const float* __restrict__ V, float* __restrict__ out, long E) {
    long e = ((long)blockIdx.x * blockDim.x + threadIdx.x) >> 5;
    int lane = threadIdx.x & 31;
    if (e >= E) return;
    long r = rows[e], c = cols[e];
    int h = lane >> 3;
    float a = expAtt[e * HEADS + h] / (attNorm[r * HEADS + h] + 1e-8f);
    float2 v = *(const float2*)(V + c * D + lane * 2);
    atomicAdd(out + r * D + lane * 2,     a * v.x);
    atomicAdd(out + r * D + lane * 2 + 1, a * v.y);
}

extern "C" void kernel_launch(void* const* d_in, const int* in_sizes, int n_in,
                              void* d_out, int out_size, void* d_ws, size_t ws_size,
                              hipStream_t stream) {
    const float* uE   = (const float*)d_in[0];
    const float* iE   = (const float*)d_in[1];
    const float* qT   = (const float*)d_in[2];
    const float* kT   = (const float*)d_in[3];
    const float* vT   = (const float*)d_in[4];
    const float* eV   = (const float*)d_in[5];
    const int*   eR   = (const int*)d_in[6];
    const int*   eC   = (const int*)d_in[7];
    const int*   dR   = (const int*)d_in[8];
    const int*   dC   = (const int*)d_in[9];
    float* out = (float*)d_out;

    const long NU = in_sizes[0] / D;
    const long NI = in_sizes[1] / D;
    const long N  = NU + NI;
    const long E  = in_sizes[5];

    // Workspace layout (floats):
    //   bufA (emb1, later Q) | bufB (emb2) | bufK | bufV | attNorm | expAtt
    float* bufA    = (float*)d_ws;           // N*D
    float* bufB    = bufA + N * D;           // N*D
    float* bufK    = bufB + N * D;           // N*D
    float* bufV    = bufK + N * D;           // N*D
    float* attNorm = bufV + N * D;           // N*HEADS
    float* expAtt  = attNorm + N * HEADS;    // E*HEADS

    const long total4 = N * D / 4;
    const int vecBlocks  = (int)((total4 + 255) / 256);
    const int edgeBlocks = (int)((E + 7) / 8);     // 8 waves / block
    const long numTiles  = (N + 31) / 32;
    const int qkvBlocks  = (int)(numTiles < 1536 ? numTiles : 1536);

    // out = emb0
    concat_kernel<<<vecBlocks, 256, 0, stream>>>(uE, iE, out, NU * D / 4, total4);

    // emb1 = gcn(emb0); out += emb1
    hipMemsetAsync(bufA, 0, (size_t)N * D * sizeof(float), stream);
    spmm_kernel<<<edgeBlocks, 256, 0, stream>>>(eV, eR, eC, out, bufA, E);
    add_kernel<<<vecBlocks, 256, 0, stream>>>(out, bufA, total4);

    // emb2 = gcn(emb1); out += emb2
    hipMemsetAsync(bufB, 0, (size_t)N * D * sizeof(float), stream);
    spmm_kernel<<<edgeBlocks, 256, 0, stream>>>(eV, eR, eC, bufA, bufB, E);
    add_kernel<<<vecBlocks, 256, 0, stream>>>(out, bufB, total4);

    // Q,K,V = emb2 @ {qT,kT,vT}  (persistent WMMA; Q overwrites bufA)
    qkv_wmma_kernel<<<qkvBlocks, 256, 0, stream>>>(bufB, qT, kT, vT,
                                                   bufA, bufK, bufV, N, numTiles);

    // attention scores + row-softmax normalization + aggregation into out
    hipMemsetAsync(attNorm, 0, (size_t)N * HEADS * sizeof(float), stream);
    att_score_kernel<<<edgeBlocks, 256, 0, stream>>>(dR, dC, bufA, bufK,
                                                     attNorm, expAtt, E);
    att_aggr_kernel<<<edgeBlocks, 256, 0, stream>>>(dR, dC, expAtt, attNorm,
                                                    bufV, out, E);
}
